// RNNModel_81286551044707
// MI455X (gfx1250) — matrix operand order repaired and checked by reading, not compiled, not run on hardware
//
#include <hip/hip_runtime.h>

typedef __attribute__((ext_vector_type(16))) _Float16 v16h;
typedef __attribute__((ext_vector_type(8)))  float    v8f;

// Global-address-space views (forced via inttoptr, which is always legal and
// guarantees global_load/global_store instruction selection).
typedef __attribute__((address_space(1))) _Float16 g_half;
typedef __attribute__((address_space(1))) v16h     g_v16h;
typedef __attribute__((address_space(1))) float    g_float;

__device__ __forceinline__ const g_half* ghp_c(const _Float16* p) {
  return (const g_half*)(unsigned long long)p;
}
__device__ __forceinline__ g_half* ghp(_Float16* p) {
  return (g_half*)(unsigned long long)p;
}
__device__ __forceinline__ const g_float* gfp_c(const float* p) {
  return (const g_float*)(unsigned long long)p;
}
__device__ __forceinline__ g_float* gfp(float* p) {
  return (g_float*)(unsigned long long)p;
}

#define BDIM 256
#define BN   16384
#define TT   5

__device__ __forceinline__ v8f vzero8() {
  v8f z = {0.f,0.f,0.f,0.f,0.f,0.f,0.f,0.f};
  return z;
}

// ---------------------------------------------------------------------------
// Packed fragment layouts (f16), both stored as 16 halves (32B) per lane,
// 32 lanes (512 halves) per 16x32 / 32x16 tile, tiles contiguous.
//
// A-image of X[M,Kc] (row-major):  tile (mt,kt), lane l, half j:
//   row  = mt*16 + (l&15)
//   col  = kt*32 + (l>=16 ? 8 : 0) + (j<8 ? j : 8+j)   (ISA 16-bit A 16x32)
//   addr = ((mt*(Kc/32)+kt)*32 + l)*16 + j
__device__ __forceinline__ void store_packA(g_half* pA, int Kc,
                                            int row, int col, float v) {
  int mt = row >> 4, kt = col >> 5, c = col & 31;
  int laneHalf = (c >> 3) & 1;
  int j = (c & 7) + ((c >> 4) << 3);
  int lane = (row & 15) + (laneHalf << 4);
  size_t idx = (((size_t)mt * (Kc >> 5) + kt) * 32 + lane) * 16 + j;
  pA[idx] = (_Float16)v;
}

// B-image of W[K,N] (row-major): tile t = kt*(N/16)+nt, lane l:
//   col = nt*16 + (l&15); kbase = kt*32 + (l>>4)*8; half j -> W[kbase+j+(j>=8?8:0), col]
__global__ void pack_b(const float* __restrict__ W, _Float16* __restrict__ out,
                       int K, int N) {
  const g_float* Wg = gfp_c(W);
  g_half* og = ghp(out);
  int tile = blockIdx.x;
  int lane = threadIdx.x;
  int ntiles = N >> 4;
  int kt = tile / ntiles, nt = tile - kt * ntiles;
  int col = nt * 16 + (lane & 15);
  int kbase = kt * 32 + ((lane >> 4) << 3);
  g_half* dst = og + ((size_t)tile * 32 + lane) * 16;
#pragma unroll
  for (int j = 0; j < 16; ++j)
    dst[j] = (_Float16)Wg[(size_t)(kbase + j + (j >= 8 ? 8 : 0)) * N + col];
}

// ---------------------------------------------------------------------------
// Core K-loop on pre-packed fragments (all-global, immediate-offset loads).
// ---------------------------------------------------------------------------
template<int MT, int NT, int NN>
__device__ __forceinline__ void gemm_core(
    const _Float16* Aq, const _Float16* Wq, int Kp,
    int mtile0, int wave, int lane, v8f (&acc)[MT][NT])
{
  constexpr int    NTILES = NN >> 4;
  constexpr size_t BSTEP  = (size_t)NTILES * 512;   // halves per K-step
  const int nkt = Kp >> 5;

  const g_half* ap[MT];
#pragma unroll
  for (int mi = 0; mi < MT; ++mi)
    ap[mi] = ghp_c(Aq) + (((size_t)(mtile0 + mi) * nkt) * 32 + lane) * 16;
  const g_half* bp[NT];
#pragma unroll
  for (int j = 0; j < NT; ++j)
    bp[j] = ghp_c(Wq) + (((size_t)(wave * NT + j)) * 32 + lane) * 16;

  for (int kt = 0; kt < nkt; kt += 4) {
#pragma unroll
    for (int u = 0; u < 4; ++u) {
      v16h a[MT];
#pragma unroll
      for (int mi = 0; mi < MT; ++mi)
        a[mi] = *(const g_v16h*)(ap[mi] + (size_t)u * 512);
#pragma unroll
      for (int j = 0; j < NT; ++j) {
        v16h b = *(const g_v16h*)(bp[j] + (size_t)u * BSTEP);
#pragma unroll
        for (int mi = 0; mi < MT; ++mi)
          acc[mi][j] = __builtin_amdgcn_wmma_f32_16x16x32_f16(
              false, a[mi], false, b, (short)0, acc[mi][j], false, false);
      }
    }
#pragma unroll
    for (int mi = 0; mi < MT; ++mi) ap[mi] += 4 * 512;
#pragma unroll
    for (int j = 0; j < NT; ++j) bp[j] += 4 * BSTEP;
  }
}

// ---------------------------------------------------------------------------
// GEMM: C[M,N] = act( A@W + A2@W2 + bias + res ), optional f32 out C and
// packed-f16 A-image out Cp. Block = 8 waves covering MT m-tiles; each wave
// owns NT n-tiles. No LDS, no barriers; WMMA-dense inner loop.
// ---------------------------------------------------------------------------
template<int MT, int NT, int NN>
__global__ __launch_bounds__(BDIM) void gemm16p(
    const _Float16* __restrict__ Ap,  const _Float16* __restrict__ Wp,  int K,
    const _Float16* __restrict__ Ap2, const _Float16* __restrict__ Wp2, int K2,
    const float* __restrict__ bias, const float* __restrict__ res,
    float* __restrict__ C, _Float16* __restrict__ Cp,
    int M, int act)
{
  const int tid  = threadIdx.x;
  const int wave = tid >> 5, lane = tid & 31;
  const int mtile0 = blockIdx.x * MT;

  v8f acc[MT][NT];
#pragma unroll
  for (int mi = 0; mi < MT; ++mi)
#pragma unroll
    for (int j = 0; j < NT; ++j) acc[mi][j] = vzero8();

  gemm_core<MT, NT, NN>(Ap, Wp, K, mtile0, wave, lane, acc);
  if (Ap2 != nullptr)
    gemm_core<MT, NT, NN>(Ap2, Wp2, K2, mtile0, wave, lane, acc);

  // C/D layout: VGPR rr -> M = rr + 8*(lane>=16), N = tile*16 + (lane&15)
  const g_float* biasg = gfp_c(bias);
  const g_float* resg  = gfp_c(res);
  g_float* Cg  = gfp(C);
  g_half*  Cpg = ghp(Cp);
  const int rbase = ((lane >> 4) << 3);
#pragma unroll
  for (int mi = 0; mi < MT; ++mi) {
#pragma unroll
    for (int j = 0; j < NT; ++j) {
      int ncol = (wave * NT + j) * 16 + (lane & 15);
      float bv = bias ? biasg[ncol] : 0.f;
#pragma unroll
      for (int rr = 0; rr < 8; ++rr) {
        int row = (mtile0 + mi) * 16 + rbase + rr;
        size_t idx = (size_t)row * NN + ncol;
        float v = acc[mi][j][rr] + bv;
        if (res) v += resg[idx];
        if (act) v = tanhf(v);
        if (C)  Cg[idx] = v;
        if (Cp) store_packA(Cpg, NN, row, ncol, v);
      }
    }
  }
}

// ---------------------------------------------------------------------------
// Elementwise producers (f32 + packed-f16 A-image outputs)
// ---------------------------------------------------------------------------
__global__ void maskfill_kernel(float* __restrict__ out, _Float16* __restrict__ pA,
                                const float* __restrict__ seq,
                                const unsigned char* __restrict__ mask,
                                const float* __restrict__ imp, int n) {
  int i = blockIdx.x * blockDim.x + threadIdx.x;
  if (i >= n) return;
  float impv = imp ? imp[i] : 0.f;
  float v = mask[i] ? impv : seq[i];
  out[i] = v;
  store_packA(ghp(pA), 256, i >> 8, i & 255, v);
}

__global__ void petfill_kernel(float* __restrict__ out, _Float16* __restrict__ pA,
                               const float* __restrict__ seq,
                               const unsigned char* __restrict__ mask,
                               const float* __restrict__ imp,
                               const float* __restrict__ est,
                               float alpha, float beta, int n) {
  int i = blockIdx.x * blockDim.x + threadIdx.x;
  if (i >= n) return;
  float fill = imp ? (alpha * imp[i] + beta * est[i]) : est[i];
  float v = mask[i] ? fill : seq[i];
  out[i] = v;
  store_packA(ghp(pA), 256, i >> 8, i & 255, v);
}

// cat[b,0:256]=a, cat[b,256:512]=c ; emits [B,512] image and [2B,256] image
__global__ void concat_kernel(float* __restrict__ cat,
                              _Float16* __restrict__ p512,
                              _Float16* __restrict__ p256,
                              const float* __restrict__ a,
                              const float* __restrict__ c, int n) {
  int i = blockIdx.x * blockDim.x + threadIdx.x;   // over B*512
  if (i >= n) return;
  int b = i >> 9, d = i & 511;
  float v = (d < 256) ? a[(size_t)b * 256 + d] : c[(size_t)b * 256 + (d - 256)];
  cat[i] = v;
  store_packA(ghp(p512), 512, b, d, v);
  store_packA(ghp(p256), 256, (b << 1) | (d >> 8), d & 255, v);
}

// ---------------------------------------------------------------------------
// MHA1 core: L=2, 8 heads, dk=32. One thread per (b,h).
// ---------------------------------------------------------------------------
__global__ void attn1_kernel(const float* __restrict__ q,
                             const float* __restrict__ k,
                             const float* __restrict__ v,
                             float* __restrict__ attn_out,   // [B,8,2,2]
                             float* __restrict__ ctx,        // [2B,256]
                             _Float16* __restrict__ ctxp,
                             int Bn) {
  int t = blockIdx.x * blockDim.x + threadIdx.x;
  if (t >= Bn * 8) return;
  int b = t >> 3, h = t & 7;
  size_t r0 = (size_t)(2 * b) * 256 + h * 32;
  size_t r1 = (size_t)(2 * b + 1) * 256 + h * 32;
  float s00 = 0, s01 = 0, s10 = 0, s11 = 0;
#pragma unroll 8
  for (int d = 0; d < 32; ++d) {
    float q0 = q[r0 + d], q1 = q[r1 + d];
    float k0 = k[r0 + d], k1 = k[r1 + d];
    s00 += q0 * k0; s01 += q0 * k1; s10 += q1 * k0; s11 += q1 * k1;
  }
  const float sc = 0.17677669529663687f;  // 1/sqrt(32)
  s00 *= sc; s01 *= sc; s10 *= sc; s11 *= sc;
  float m0 = fmaxf(s00, s01), m1 = fmaxf(s10, s11);
  float e00 = expf(s00 - m0), e01 = expf(s01 - m0);
  float e10 = expf(s10 - m1), e11 = expf(s11 - m1);
  float d0 = 1.f / (e00 + e01), d1 = 1.f / (e10 + e11);
  float a00 = e00 * d0, a01 = e01 * d0, a10 = e10 * d1, a11 = e11 * d1;
  float* ao = attn_out + (size_t)t * 4;
  ao[0] = a00; ao[1] = a01; ao[2] = a10; ao[3] = a11;
  g_half* cg = ghp(ctxp);
#pragma unroll 8
  for (int d = 0; d < 32; ++d) {
    float v0 = v[r0 + d], v1 = v[r1 + d];
    float c0 = a00 * v0 + a01 * v1;
    float c1 = a10 * v0 + a11 * v1;
    ctx[r0 + d] = c0;
    ctx[r1 + d] = c1;
    store_packA(cg, 256, 2 * b,     h * 32 + d, c0);
    store_packA(cg, 256, 2 * b + 1, h * 32 + d, c1);
  }
}

// ---------------------------------------------------------------------------
// MHA2 core: L=5, 8 heads, dk=64. One thread per (b,h).
// ---------------------------------------------------------------------------
__global__ void attn2_kernel(const float* __restrict__ q,
                             const float* __restrict__ k,
                             const float* __restrict__ v,
                             float* __restrict__ attn_out,   // [B,8,5,5]
                             float* __restrict__ ctx,        // [5B,512]
                             _Float16* __restrict__ ctxp,
                             int Bn) {
  int t = blockIdx.x * blockDim.x + threadIdx.x;
  if (t >= Bn * 8) return;
  int b = t >> 3, h = t & 7;
  const size_t base = (size_t)b * 5 * 512 + h * 64;
  float s[5][5];
#pragma unroll
  for (int i = 0; i < 5; ++i)
#pragma unroll
    for (int j = 0; j < 5; ++j) s[i][j] = 0.f;
  for (int d = 0; d < 64; ++d) {
    float qv[5], kv[5];
#pragma unroll
    for (int i = 0; i < 5; ++i) {
      qv[i] = q[base + (size_t)i * 512 + d];
      kv[i] = k[base + (size_t)i * 512 + d];
    }
#pragma unroll
    for (int i = 0; i < 5; ++i)
#pragma unroll
      for (int j = 0; j < 5; ++j) s[i][j] += qv[i] * kv[j];
  }
  const float sc = 0.125f;  // 1/sqrt(64)
  float a[5][5];
#pragma unroll
  for (int i = 0; i < 5; ++i) {
    float mx = -1e30f;
#pragma unroll
    for (int j = 0; j < 5; ++j) { s[i][j] *= sc; mx = fmaxf(mx, s[i][j]); }
    float den = 0.f;
#pragma unroll
    for (int j = 0; j < 5; ++j) { a[i][j] = expf(s[i][j] - mx); den += a[i][j]; }
    float rden = 1.f / den;
#pragma unroll
    for (int j = 0; j < 5; ++j) {
      a[i][j] *= rden;
      attn_out[(size_t)t * 25 + i * 5 + j] = a[i][j];
    }
  }
  g_half* cg = ghp(ctxp);
  for (int d = 0; d < 64; ++d) {
    float vv[5];
#pragma unroll
    for (int j = 0; j < 5; ++j) vv[j] = v[base + (size_t)j * 512 + d];
#pragma unroll
    for (int i = 0; i < 5; ++i) {
      float o = 0.f;
#pragma unroll
      for (int j = 0; j < 5; ++j) o += a[i][j] * vv[j];
      ctx[base + (size_t)i * 512 + d] = o;
      store_packA(cg, 512, b * 5 + i, h * 64 + d, o);
    }
  }
}

// ---------------------------------------------------------------------------
// LayerNorm over last dim D. One 256-thread block per row.
// remap=1 (MHA1): row=2b+tok -> out row R=b*5+step, col=tok*256+d, width 512,
//                 plus packed image of the [5B,512] matrix.
// ---------------------------------------------------------------------------
template<int D>
__global__ __launch_bounds__(BDIM) void ln_kernel(
    const float* __restrict__ x, const float* __restrict__ g,
    const float* __restrict__ bb, float* __restrict__ out,
    _Float16* __restrict__ pA, int remap, int step) {
  __shared__ float s1[BDIM], s2[BDIM];
  const int row = blockIdx.x, tid = threadIdx.x;
  const int PER = D / BDIM;  // 1 or 2
  const float* xr = x + (size_t)row * D;
  float vals[PER];
  float lsum = 0.f, lsq = 0.f;
#pragma unroll
  for (int j = 0; j < PER; ++j) {
    float v = xr[tid + j * BDIM];
    vals[j] = v; lsum += v; lsq += v * v;
  }
  s1[tid] = lsum; s2[tid] = lsq;
  __syncthreads();
  for (int st = BDIM / 2; st > 0; st >>= 1) {
    if (tid < st) { s1[tid] += s1[tid + st]; s2[tid] += s2[tid + st]; }
    __syncthreads();
  }
  float mean = s1[0] * (1.f / D);
  float var  = s2[0] * (1.f / D) - mean * mean;
  float inv  = rsqrtf(var + 1e-6f);
  g_half* pg = ghp(pA);
#pragma unroll
  for (int j = 0; j < PER; ++j) {
    int d = tid + j * BDIM;
    float o = g[d] * (vals[j] - mean) * inv + bb[d];
    if (remap) {
      int b = row >> 1, tok = row & 1;
      int R = b * 5 + step, col = tok * 256 + d;
      out[(size_t)R * 512 + col] = o;
      if (pA) store_packA(pg, 512, R, col, o);
    } else {
      out[(size_t)row * D + d] = o;
      if (pA) store_packA(pg, D, row, d, o);
    }
  }
}

// ---------------------------------------------------------------------------
// Tiny FC heads
// ---------------------------------------------------------------------------
__global__ void fc3_kernel(const float* __restrict__ x, int xstride,
                           const float* __restrict__ W,
                           const float* __restrict__ bias,
                           float* __restrict__ out, int Kdim,
                           int do_softmax, int Bn) {
  int b = blockIdx.x * blockDim.x + threadIdx.x;
  if (b >= Bn) return;
  const float* xr = x + (size_t)b * xstride;
  float s0 = bias[0], s1 = bias[1], s2 = bias[2];
  for (int kk = 0; kk < Kdim; ++kk) {
    float v = xr[kk];
    s0 += v * W[kk * 3 + 0];
    s1 += v * W[kk * 3 + 1];
    s2 += v * W[kk * 3 + 2];
  }
  if (do_softmax) {
    float m = fmaxf(s0, fmaxf(s1, s2));
    float e0 = expf(s0 - m), e1 = expf(s1 - m), e2 = expf(s2 - m);
    float d = 1.f / (e0 + e1 + e2);
    s0 = e0 * d; s1 = e1 * d; s2 = e2 * d;
  }
  out[(size_t)b * 3 + 0] = s0;
  out[(size_t)b * 3 + 1] = s1;
  out[(size_t)b * 3 + 2] = s2;
}

__global__ void trans_kernel(const float* __restrict__ x,
                             const float* __restrict__ W,
                             const float* __restrict__ bias,
                             float* __restrict__ out, int Bn) {
  int b = blockIdx.x * blockDim.x + threadIdx.x;
  if (b >= Bn) return;
  const float* xr = x + (size_t)b * 2560;
  float s0 = bias[0], s1 = bias[1];
  for (int kk = 0; kk < 2560; ++kk) {
    float v = xr[kk];
    s0 += v * W[kk * 2 + 0];
    s1 += v * W[kk * 2 + 1];
  }
  float m = fmaxf(s0, s1);
  float e0 = expf(s0 - m), e1 = expf(s1 - m);
  float d = 1.f / (e0 + e1);
  out[(size_t)b * 2 + 0] = e0 * d;
  out[(size_t)b * 2 + 1] = e1 * d;
}

// ---------------------------------------------------------------------------
extern "C" void kernel_launch(void* const* d_in, const int* in_sizes, int n_in,
                              void* d_out, int out_size, void* d_ws, size_t ws_size,
                              hipStream_t stream) {
  (void)in_sizes; (void)n_in; (void)out_size; (void)ws_size;
  const int Bn = BN;
  const float ALPHA = 0.5f, BETA = 0.5f;

  const float* mri_seq = (const float*)d_in[0];
  const float* pet_seq = (const float*)d_in[1];
  const unsigned char* mri_mask = (const unsigned char*)d_in[2];
  const unsigned char* pet_mask = (const unsigned char*)d_in[3];
  const float* b_mri0 = (const float*)d_in[6];
  const float* b_mri1 = (const float*)d_in[9];
  const float* b_pet0 = (const float*)d_in[12];
  const float* b_pet1 = (const float*)d_in[15];
  const float* b_p11  = (const float*)d_in[17];
  const float* b_p12  = (const float*)d_in[19];
  const float* W_h2c  = (const float*)d_in[20];
  const float* b_h2c  = (const float*)d_in[21];
  const float* b_h2m  = (const float*)d_in[23];
  const float* b_p2   = (const float*)d_in[25];
  const float* g1     = (const float*)d_in[30];
  const float* bln1   = (const float*)d_in[31];
  const float* g2     = (const float*)d_in[36];
  const float* bln2   = (const float*)d_in[37];
  const float* W_tr   = (const float*)d_in[38];
  const float* b_tr   = (const float*)d_in[39];

  // ---- pack weights into WMMA B-fragment layout (f16) ----
  enum { PW_WXM0, PW_WHM0, PW_WXM1, PW_WHM1, PW_WXP0, PW_WHP0, PW_WXP1, PW_WHP1,
         PW_P11, PW_P12, PW_H2M, PW_P2, PW_Q1, PW_K1, PW_V1, PW_O1,
         PW_Q2, PW_K2, PW_V2, PW_O2, NPW };
  static const int pwIdx[NPW] = {4,5,7,8,10,11,13,14,16,18,22,24,26,27,28,29,32,33,34,35};
  static const int pwK[NPW]   = {256,256,256,256,256,256,256,256,256,128,512,512,
                                 256,256,256,256,512,512,512,512};
  static const int pwN[NPW]   = {256,256,256,256,256,256,256,256,128,256,256,256,
                                 256,256,256,256,512,512,512,512};
  char* wsb = (char*)d_ws;
  size_t byteOff = 0;
  auto hbuf = [&](size_t nh) {
    _Float16* p = (_Float16*)(wsb + byteOff);
    byteOff = (byteOff + nh * 2 + 255) & ~(size_t)255;
    return p;
  };
  auto fbuf = [&](size_t nflt) {
    float* p = (float*)(wsb + byteOff);
    byteOff = (byteOff + nflt * 4 + 255) & ~(size_t)255;
    return p;
  };

  _Float16* packPtr[NPW];
  for (int i = 0; i < NPW; ++i) packPtr[i] = hbuf((size_t)pwK[i] * pwN[i]);
  for (int i = 0; i < NPW; ++i) {
    int tiles = (pwK[i] / 32) * (pwN[i] / 16);
    pack_b<<<tiles, 32, 0, stream>>>((const float*)d_in[pwIdx[i]], packPtr[i],
                                     pwK[i], pwN[i]);
  }

  // ---- f32 working buffers ----
  float* x_mri  = fbuf((size_t)Bn * 256);
  float* x_pet  = fbuf((size_t)Bn * 256);
  float* hm1f   = fbuf((size_t)Bn * 256);
  float* hp1f   = fbuf((size_t)Bn * 256);
  float* cat_h  = fbuf((size_t)Bn * 512);       // == ht_set [2B,256]
  float* h_out  = fbuf((size_t)Bn * 2560);      // [B,5,512]
  float* big    = fbuf((size_t)4 * Bn * 2560);  // shared MHA1/MHA2 pool
  float* q2 = big, *k2 = big + (size_t)Bn * 2560,
       * v2 = big + (size_t)2 * Bn * 2560, *ctx2 = big + (size_t)3 * Bn * 2560;
  float* q1 = big, *k1 = big + (size_t)Bn * 512, *v1 = big + (size_t)2 * Bn * 512,
       * ctx1 = big + (size_t)3 * Bn * 512, *o1 = big + (size_t)4 * Bn * 512;
  float* o2 = q2;        // reuse after attn2 consumed q2
  float* h_out2 = k2;    // reuse after attn2 consumed k2

  // ---- packed-f16 A-images of activations ----
  _Float16* xmri_p   = hbuf((size_t)Bn * 256);
  _Float16* xpet_p   = hbuf((size_t)Bn * 256);
  _Float16* hm0p[2]  = { hbuf((size_t)Bn * 256), hbuf((size_t)Bn * 256) };
  _Float16* hm1p[2]  = { hbuf((size_t)Bn * 256), hbuf((size_t)Bn * 256) };
  _Float16* hp0p[2]  = { hbuf((size_t)Bn * 256), hbuf((size_t)Bn * 256) };
  _Float16* hp1p[2]  = { hbuf((size_t)Bn * 256), hbuf((size_t)Bn * 256) };
  _Float16* tmp128p  = hbuf((size_t)Bn * 128);
  _Float16* catp512  = hbuf((size_t)Bn * 512);
  _Float16* catp256  = hbuf((size_t)Bn * 512);
  _Float16* ctx1p    = hbuf((size_t)Bn * 512);
  _Float16* houtp    = hbuf((size_t)Bn * 2560);
  _Float16* ctx2p    = hbuf((size_t)Bn * 2560);

  // ---- d_out section offsets (flat f32, tuple return order) ----
  float* out = (float*)d_out;
  const size_t oc  = 0;                              // out_cat [10,B,3]
  const size_t om  = oc  + (size_t)10 * Bn * 3;      // out_mri [4,B,256]
  const size_t ops = om  + (size_t)4 * Bn * 256;     // pet_s   [5,B,256]
  const size_t opf = ops + (size_t)5 * Bn * 256;     // pet_f   [4,B,256]
  const size_t otr = opf + (size_t)4 * Bn * 256;     // trans   [B,2]
  const size_t oa1 = otr + (size_t)Bn * 2;           // attns1  [5,B,8,2,2]
  const size_t oa2 = oa1 + (size_t)5 * Bn * 32;      // attn2   [B,8,5,5]

  auto gemm = [&](const _Float16* A, const _Float16* W, int K,
                  const _Float16* A2, const _Float16* W2, int K2,
                  const float* bias, const float* res,
                  float* C, _Float16* Cp, int M, int N, int act) {
    dim3 blk(BDIM);
    if (N == 128) {
      dim3 g(M / 64);   // MT=4
      gemm16p<4,1,128><<<g, blk, 0, stream>>>(A, W, K, A2, W2, K2, bias, res, C, Cp, M, act);
    } else if (N == 256) {
      dim3 g(M / 64);   // MT=4
      gemm16p<4,2,256><<<g, blk, 0, stream>>>(A, W, K, A2, W2, K2, bias, res, C, Cp, M, act);
    } else {
      dim3 g(M / 32);   // MT=2
      gemm16p<2,4,512><<<g, blk, 0, stream>>>(A, W, K, A2, W2, K2, bias, res, C, Cp, M, act);
    }
  };

  const int nEl = Bn * 256;
  const int ewBlocks = nEl / BDIM;
  const int bhBlocks = (Bn * 8 + BDIM - 1) / BDIM;
  const int fcBlocks = (Bn + BDIM - 1) / BDIM;

  for (int i = 0; i < TT; ++i) {
    const int cur = i & 1, prv = cur ^ 1;
    const float* mri_imp = (i == 0) ? nullptr : out + om + (size_t)(i - 1) * nEl;
    const float* pet_imp = (i == 0) ? nullptr : out + opf + (size_t)(i - 1) * nEl;
    const _Float16* hm0pp = (i == 0) ? nullptr : hm0p[prv];
    const _Float16* hm1pp = (i == 0) ? nullptr : hm1p[prv];
    const _Float16* hp0pp = (i == 0) ? nullptr : hp0p[prv];
    const _Float16* hp1pp = (i == 0) ? nullptr : hp1p[prv];

    // x_mri = where(mask, mri_imp, mri_seq[i])
    maskfill_kernel<<<ewBlocks, BDIM, 0, stream>>>(
        x_mri, xmri_p, mri_seq + (size_t)i * nEl, mri_mask + (size_t)i * nEl,
        mri_imp, nEl);

    // MRI RNN cell (2 layers); layer-0 hidden kept only as packed image
    gemm(xmri_p, packPtr[PW_WXM0], 256, hm0pp, packPtr[PW_WHM0], 256,
         b_mri0, nullptr, nullptr, hm0p[cur], Bn, 256, 1);
    gemm(hm0p[cur], packPtr[PW_WXM1], 256, hm1pp, packPtr[PW_WHM1], 256,
         b_mri1, nullptr, hm1f, hm1p[cur], Bn, 256, 1);

    // pet_est1 = tanh(h1@Wp11+b)@Wp12 + b + x_mri  -> pet_s slot i
    float* pet_s_i = out + ops + (size_t)i * nEl;
    gemm(hm1p[cur], packPtr[PW_P11], 256, nullptr, nullptr, 0,
         b_p11, nullptr, nullptr, tmp128p, Bn, 128, 1);
    gemm(tmp128p, packPtr[PW_P12], 128, nullptr, nullptr, 0,
         b_p12, x_mri, pet_s_i, nullptr, Bn, 256, 0);

    // x_pet = where(mask, i==0 ? est : a*imp + b*est, pet_seq[i])
    petfill_kernel<<<ewBlocks, BDIM, 0, stream>>>(
        x_pet, xpet_p, pet_seq + (size_t)i * nEl, pet_mask + (size_t)i * nEl,
        pet_imp, pet_s_i, ALPHA, BETA, nEl);

    // PET RNN cell
    gemm(xpet_p, packPtr[PW_WXP0], 256, hp0pp, packPtr[PW_WHP0], 256,
         b_pet0, nullptr, nullptr, hp0p[cur], Bn, 256, 1);
    gemm(hp0p[cur], packPtr[PW_WXP1], 256, hp1pp, packPtr[PW_WHP1], 256,
         b_pet1, nullptr, hp1f, hp1p[cur], Bn, 256, 1);

    // cat_h = [mri_h | pet_h]; also [2B,256] token matrix (two packed images)
    concat_kernel<<<(Bn * 512) / BDIM, BDIM, 0, stream>>>(
        cat_h, catp512, catp256, hm1f, hp1f, Bn * 512);

    // out_cat[2i] = cat_h @ W_h2c + b_h2c
    fc3_kernel<<<fcBlocks, BDIM, 0, stream>>>(
        cat_h, 512, W_h2c, b_h2c, out + oc + (size_t)(2 * i) * Bn * 3, 512, 0, Bn);

    if (i < TT - 1) {
      gemm(catp512, packPtr[PW_H2M], 512, nullptr, nullptr, 0,
           b_h2m, x_mri, out + om + (size_t)i * nEl, nullptr, Bn, 256, 0);
      gemm(catp512, packPtr[PW_P2], 512, nullptr, nullptr, 0,
           b_p2, x_pet, out + opf + (size_t)i * nEl, nullptr, Bn, 256, 0);
    }

    // MHA1 projections over [2B,256]
    gemm(catp256, packPtr[PW_Q1], 256, nullptr, nullptr, 0,
         nullptr, nullptr, q1, nullptr, 2 * Bn, 256, 0);
    gemm(catp256, packPtr[PW_K1], 256, nullptr, nullptr, 0,
         nullptr, nullptr, k1, nullptr, 2 * Bn, 256, 0);
    gemm(catp256, packPtr[PW_V1], 256, nullptr, nullptr, 0,
         nullptr, nullptr, v1, nullptr, 2 * Bn, 256, 0);
    attn1_kernel<<<bhBlocks, BDIM, 0, stream>>>(
        q1, k1, v1, out + oa1 + (size_t)i * Bn * 32, ctx1, ctx1p, Bn);
    gemm(ctx1p, packPtr[PW_O1], 256, nullptr, nullptr, 0,
         nullptr, cat_h, o1, nullptr, 2 * Bn, 256, 0);
    ln_kernel<256><<<2 * Bn, BDIM, 0, stream>>>(o1, g1, bln1, h_out, houtp, 1, i);

    // out_cat[2i+1] = softmax(ht_new.reshape(B,512) @ W_h2c + b)
    fc3_kernel<<<fcBlocks, BDIM, 0, stream>>>(
        h_out + (size_t)i * 512, 2560, W_h2c, b_h2c,
        out + oc + (size_t)(2 * i + 1) * Bn * 3, 512, 1, Bn);
  }

  // MHA2 over [5B, 512]
  gemm(houtp, packPtr[PW_Q2], 512, nullptr, nullptr, 0,
       nullptr, nullptr, q2, nullptr, 5 * Bn, 512, 0);
  gemm(houtp, packPtr[PW_K2], 512, nullptr, nullptr, 0,
       nullptr, nullptr, k2, nullptr, 5 * Bn, 512, 0);
  gemm(houtp, packPtr[PW_V2], 512, nullptr, nullptr, 0,
       nullptr, nullptr, v2, nullptr, 5 * Bn, 512, 0);
  attn2_kernel<<<bhBlocks, BDIM, 0, stream>>>(q2, k2, v2, out + oa2, ctx2, ctx2p, Bn);
  gemm(ctx2p, packPtr[PW_O2], 512, nullptr, nullptr, 0,
       nullptr, h_out, o2, nullptr, 5 * Bn, 512, 0);
  ln_kernel<512><<<5 * Bn, BDIM, 0, stream>>>(o2, g2, bln2, h_out2, nullptr, 0, 0);

  // trans_out = softmax(h_out2.reshape(B,2560) @ W_trans + b, axis=-1)
  trans_kernel<<<fcBlocks, BDIM, 0, stream>>>(h_out2, W_tr, b_tr, out + otr, Bn);
}
